// GraphAttentionNetwork_51161650430527
// MI455X (gfx1250) — matrix operand order repaired and checked by reading, not compiled, not run on hardware
//
#include <hip/hip_runtime.h>
#include <stdint.h>

// ---------------- problem constants (fixed by the reference) ----------------
#define NNODES   10000
#define FIN      128
#define HID      128
#define HEADS    4
#define C1       (HEADS*HID)   // 512
#define OUTD     256
#define NEG_SLOPE 0.2f

typedef __attribute__((ext_vector_type(16))) _Float16     v16h;
typedef __attribute__((ext_vector_type(8)))  float        v8f;
typedef __attribute__((ext_vector_type(8)))  unsigned int v8u;

// ---------------- workspace layout (bytes) ----------------
// layer-2 buffers alias layer-1 buffers that are dead by then.
#define OFF_XW1     0u            // f32 [10000,512] = 20,480,000   (xw2 aliases)
#define OFF_OUT1    20480000u     // f32 [10000,512] = 20,480,000   (out2 aliases)
#define OFF_APACK1  40960000u     // f16 packed emb  =  2,560,000
#define OFF_APACK2  43520000u     // f16 packed x2   = 10,240,000
#define OFF_BPACK1  53760000u     // f16 packed W1   =    131,072
#define OFF_BPACK2  53891072u     // f16 packed W2   =    262,144
#define OFF_EX1     54153216u     // f32 [170000,4]  =  2,720,000   (ex2 aliases)
#define OFF_Z1      56873216u     // f32 [10000,4]   =    160,000   (z2 aliases)
#define OFF_AS1     57033216u     // f32 [10000,4]   =    160,000   (as2 aliases)
#define OFF_AD1     57193216u     // f32 [10000,4]   =    160,000   (ad2 aliases)
#define OFF_GSUM    57353216u     // f32 [256]       =      1,024
// total ~57.4 MB

static __device__ __forceinline__ void atomAddF(float* p, float v) {
  unsafeAtomicAdd(p, v);   // hardware global_atomic_add_f32
}

static __device__ __forceinline__ unsigned pack2h(float a, float b) {
  _Float16 ha = (_Float16)a, hb = (_Float16)b;
  unsigned short ua = __builtin_bit_cast(unsigned short, ha);
  unsigned short ub = __builtin_bit_cast(unsigned short, hb);
  return (unsigned)ua | ((unsigned)ub << 16);
}

// ---------------- generic zero fill ----------------
__global__ void k_zero(float* __restrict__ p, int n) {
  int i = blockIdx.x * blockDim.x + threadIdx.x;
  if (i < n) p[i] = 0.f;
}

// ---------------- f32 -> f16 A-fragment pack (WMMA 16x16x32 A layout) -------
// linear idx = ((mt*KS + ks)*32 + lane)*8 + r ; element pair (k,k+1) contiguous.
__global__ void k_pack_a(const float* __restrict__ A, unsigned* __restrict__ out,
                         int Mtiles, int KS, int K) {
  int idx = blockIdx.x * blockDim.x + threadIdx.x;
  int total = Mtiles * KS * 32 * 8;
  if (idx >= total) return;
  int r    = idx & 7;
  int lane = (idx >> 3) & 31;
  int t    = idx >> 8;
  int ks   = t % KS;
  int mt   = t / KS;
  int m = mt * 16 + (lane & 15);
  int k = ks * 32 + ((r < 4) ? 0 : 16) + ((lane >> 4) << 3) + ((r & 3) << 1);
  const float* row = A + (long)m * K + k;
  out[idx] = pack2h(row[0], row[1]);
}

// ---------------- f32 -> f16 B-fragment pack (WMMA 16x16x32 B layout) -------
// lane = N within tile; VGPR r halves are rows k0=ks*32+16*(r>=4)+8*(lane>>4)+2*(r&3), k0+1.
__global__ void k_pack_b(const float* __restrict__ B, unsigned* __restrict__ out,
                         int Ntiles, int KS, int N) {
  int idx = blockIdx.x * blockDim.x + threadIdx.x;
  int total = Ntiles * KS * 32 * 8;
  if (idx >= total) return;
  int r    = idx & 7;
  int lane = (idx >> 3) & 31;
  int t    = idx >> 8;
  int ks   = t % KS;
  int nt   = t / KS;
  int n  = nt * 16 + (lane & 15);
  int k0 = ks * 32 + ((r < 4) ? 0 : 16) + ((lane >> 4) << 3) + ((r & 3) << 1);
  out[idx] = pack2h(B[(long)k0 * N + n], B[(long)(k0 + 1) * N + n]);
}

// ---------------- 16x16 f32 C-tile store (WMMA C/D layout) ------------------
static __device__ __forceinline__ void store_tile(float* __restrict__ C, int N,
                                                  int rbase, int col, v8f acc) {
#pragma unroll
  for (int v = 0; v < 8; ++v)
    C[(long)(rbase + v) * N + col] = acc[v];
}

// ---------------- WMMA GEMM: one wave per 16x64 C block ---------------------
// 4 independent accumulator chains per wave -> no back-to-back D->C stall on
// the matrix pipe; each A fragment feeds 4 WMMAs (4x less A traffic).
__global__ void k_gemm4(const v8u* __restrict__ Ap, const v8u* __restrict__ Bp,
                        float* __restrict__ C, int Mtiles, int NtB, int KS, int N) {
  int wid  = (blockIdx.x * blockDim.x + threadIdx.x) >> 5;
  int lane = threadIdx.x & 31;
  if (wid >= Mtiles * NtB) return;   // wave-uniform: EXEC stays all-ones
  int mt  = wid % Mtiles;
  int ntb = wid / Mtiles;
  const int bstride = KS * 32;       // v8u elements per n-tile of B
  const v8u* ap = Ap + (long)mt * bstride + lane;
  const v8u* bp = Bp + (long)(ntb * 4) * bstride + lane;

  v8f acc0 = {}, acc1 = {}, acc2 = {}, acc3 = {};
  for (int ks = 0; ks < KS; ++ks) {
    v16h a  = __builtin_bit_cast(v16h, ap[ks * 32]);
    v16h b0 = __builtin_bit_cast(v16h, bp[ks * 32]);
    v16h b1 = __builtin_bit_cast(v16h, bp[bstride + ks * 32]);
    v16h b2 = __builtin_bit_cast(v16h, bp[2 * bstride + ks * 32]);
    v16h b3 = __builtin_bit_cast(v16h, bp[3 * bstride + ks * 32]);
    if (ks + 1 < KS) {               // global_prefetch_b8 for next K-step
      __builtin_prefetch((const void*)(ap + (ks + 1) * 32), 0, 3);
      __builtin_prefetch((const void*)(bp + 3 * bstride + (ks + 1) * 32), 0, 3);
    }
    acc0 = __builtin_amdgcn_wmma_f32_16x16x32_f16(false, a, false, b0, (short)0, acc0, false, false);
    acc1 = __builtin_amdgcn_wmma_f32_16x16x32_f16(false, a, false, b1, (short)0, acc1, false, false);
    acc2 = __builtin_amdgcn_wmma_f32_16x16x32_f16(false, a, false, b2, (short)0, acc2, false, false);
    acc3 = __builtin_amdgcn_wmma_f32_16x16x32_f16(false, a, false, b3, (short)0, acc3, false, false);
  }
  // C/D layout: VGPR v -> row = v + 8*(lane>>4), col = lane&15
  int rbase = mt * 16 + ((lane >> 4) << 3);
  int col0  = (ntb * 4) * 16 + (lane & 15);
  store_tile(C, N, rbase, col0,      acc0);
  store_tile(C, N, rbase, col0 + 16, acc1);
  store_tile(C, N, rbase, col0 + 32, acc2);
  store_tile(C, N, rbase, col0 + 48, acc3);
}

// ---------------- per-(node,head) attention logits: one wave each -----------
__global__ void k_alpha(const float* __restrict__ xw, const float* __restrict__ a_src,
                        const float* __restrict__ a_dst, float* __restrict__ as,
                        float* __restrict__ ad, int nNodes, int heads, int ch) {
  int wid  = (blockIdx.x * blockDim.x + threadIdx.x) >> 5;
  int lane = threadIdx.x & 31;
  if (wid >= nNodes * heads) return;
  int h = wid % heads;
  int n = wid / heads;
  const float* xr = xw + (long)n * heads * ch + (long)h * ch;
  const float* s  = a_src + h * ch;
  const float* d  = a_dst + h * ch;
  float s0 = 0.f, s1 = 0.f;
  for (int c = lane; c < ch; c += 32) {
    float x = xr[c];
    s0 += x * s[c];
    s1 += x * d[c];
  }
  for (int off = 16; off; off >>= 1) {
    s0 += __shfl_down(s0, off, 32);
    s1 += __shfl_down(s1, off, 32);
  }
  if (lane == 0) { as[wid] = s0; ad[wid] = s1; }
}

// ---------------- edge pass: ex = exp(leaky(e)), z[dst] += ex ---------------
// softmax is shift-invariant and |e| is tiny here, so the segment-max shift is
// dropped (mathematically identical result, one fewer atomic pass).
__global__ void k_edge(const int* __restrict__ ei, int E, int nNodes,
                       const float* __restrict__ as, const float* __restrict__ ad,
                       float* __restrict__ ex, float* __restrict__ z, int heads) {
  int idx = blockIdx.x * blockDim.x + threadIdx.x;
  int total = (E + nNodes) * heads;
  if (idx >= total) return;
  int h = idx % heads;
  int e = idx / heads;
  int s, d;
  if (e < E) { s = ei[e]; d = ei[E + e]; } else { s = d = e - E; }
  float v = as[s * heads + h] + ad[d * heads + h];
  v = (v > 0.f) ? v : NEG_SLOPE * v;     // leaky_relu
  float ev = __expf(v);
  ex[idx] = ev;
  atomAddF(&z[d * heads + h], ev);
}

// ---------------- edge aggregation: out[dst] += (ex/z[dst]) * xw[src] -------
__global__ void k_agg(const int* __restrict__ ei, int E, int nNodes,
                      const float* __restrict__ ex, const float* __restrict__ z,
                      const float* __restrict__ xw, float* __restrict__ out,
                      int heads, int C) {
  int e = blockIdx.x;   // one block per edge (incl. self loops)
  int s, d;
  if (e < E) { s = ei[e]; d = ei[E + e]; } else { s = d = e - E; }
  int ch = C / heads;
  const float* xr = xw + (long)s * C;
  float* orow = out + (long)d * C;
  for (int k = threadIdx.x; k < C; k += blockDim.x) {
    int h = k / ch;
    float w = ex[e * heads + h] / z[d * heads + h];
    atomAddF(&orow[k], w * xr[k]);
  }
}

// ---------------- bias + ELU fused into layer-2 A-fragment pack -------------
__global__ void k_elu_pack(const float* __restrict__ acc, const float* __restrict__ bias,
                           unsigned* __restrict__ out, int Mtiles, int KS, int K) {
  int idx = blockIdx.x * blockDim.x + threadIdx.x;
  int total = Mtiles * KS * 32 * 8;
  if (idx >= total) return;
  int r    = idx & 7;
  int lane = (idx >> 3) & 31;
  int t    = idx >> 8;
  int ks   = t % KS;
  int mt   = t / KS;
  int m = mt * 16 + (lane & 15);
  int k = ks * 32 + ((r < 4) ? 0 : 16) + ((lane >> 4) << 3) + ((r & 3) << 1);
  float f0 = acc[(long)m * K + k]     + bias[k];
  float f1 = acc[(long)m * K + k + 1] + bias[k + 1];
  f0 = (f0 > 0.f) ? f0 : (__expf(f0) - 1.f);   // ELU
  f1 = (f1 > 0.f) ? f1 : (__expf(f1) - 1.f);
  out[idx] = pack2h(f0, f1);
}

// ---------------- column sums of out2 (coalesced, partial per block) --------
__global__ void k_colsum(const float* __restrict__ out2, float* __restrict__ gsum,
                         int nNodes, int C) {
  int t = threadIdx.x;                    // t indexes column, blockDim.x == C
  float accv = 0.f;
  for (int r = blockIdx.x; r < nNodes; r += gridDim.x)
    accv += out2[(long)r * C + t];
  atomAddF(&gsum[t], accv);
}

// ---------------- mean + bias, broadcast to batch rows ----------------------
__global__ void k_writeout(const float* __restrict__ gsum, const float* __restrict__ b2,
                           float* __restrict__ out, int B, int C, float inv) {
  int k = threadIdx.x;
  if (k >= C) return;
  float v = gsum[k] * inv + b2[k];
  for (int b = 0; b < B; ++b) out[b * C + k] = v;
}

// ============================================================================
extern "C" void kernel_launch(void* const* d_in, const int* in_sizes, int n_in,
                              void* d_out, int out_size, void* d_ws, size_t ws_size,
                              hipStream_t stream) {
  const int*   ei     = (const int*)  d_in[0];
  const float* emb    = (const float*)d_in[2];
  const float* W1     = (const float*)d_in[3];
  const float* a_src1 = (const float*)d_in[4];
  const float* a_dst1 = (const float*)d_in[5];
  const float* b1     = (const float*)d_in[6];
  const float* W2     = (const float*)d_in[7];
  const float* a_src2 = (const float*)d_in[8];
  const float* a_dst2 = (const float*)d_in[9];
  const float* b2     = (const float*)d_in[10];
  float* out = (float*)d_out;

  const int E = in_sizes[0] / 2;          // 160000 real edges
  const int Etot = E + NNODES;            // + self loops
  char* ws = (char*)d_ws;

  float*    xw1    = (float*)   (ws + OFF_XW1);
  float*    out1   = (float*)   (ws + OFF_OUT1);
  unsigned* apack1 = (unsigned*)(ws + OFF_APACK1);
  unsigned* apack2 = (unsigned*)(ws + OFF_APACK2);
  unsigned* bpack1 = (unsigned*)(ws + OFF_BPACK1);
  unsigned* bpack2 = (unsigned*)(ws + OFF_BPACK2);
  float*    ex1    = (float*)   (ws + OFF_EX1);
  float*    z1     = (float*)   (ws + OFF_Z1);
  float*    as1    = (float*)   (ws + OFF_AS1);
  float*    ad1    = (float*)   (ws + OFF_AD1);
  float*    gsum   = (float*)   (ws + OFF_GSUM);
  // layer-2 aliases (regions dead by the time these are written)
  float* xw2 = xw1;
  float* out2 = out1;
  float* ex2 = ex1;
  float* z2 = z1;
  float* as2 = as1;
  float* ad2 = ad1;

  const int TB = 256;
  const int Mt  = NNODES / 16;  // 625
  const int Nt1 = C1 / 16;      // 32
  const int KS1 = FIN / 32;     // 4
  const int Nt2 = OUTD / 16;    // 16
  const int KS2 = C1 / 32;      // 16
  const int NtB1 = Nt1 / 4;     // 8  (16x64 blocks)
  const int NtB2 = Nt2 / 4;     // 4
  auto blocks = [](long n, int tb) { return (unsigned)((n + tb - 1) / tb); };

  // ---- pack f16 fragments ----
  k_pack_a<<<blocks((long)Mt*KS1*256, TB), TB, 0, stream>>>(emb, apack1, Mt, KS1, FIN);
  k_pack_b<<<blocks((long)Nt1*KS1*256, TB), TB, 0, stream>>>(W1, bpack1, Nt1, KS1, C1);
  k_pack_b<<<blocks((long)Nt2*KS2*256, TB), TB, 0, stream>>>(W2, bpack2, Nt2, KS2, OUTD);

  // ---- layer 1 ----
  k_gemm4<<<blocks((long)Mt*NtB1*32, TB), TB, 0, stream>>>((const v8u*)apack1, (const v8u*)bpack1,
                                                           xw1, Mt, NtB1, KS1, C1);
  k_alpha<<<blocks((long)NNODES*HEADS*32, TB), TB, 0, stream>>>(xw1, a_src1, a_dst1,
                                                                as1, ad1, NNODES, HEADS, HID);
  k_zero<<<blocks((long)NNODES*C1, TB), TB, 0, stream>>>(out1, NNODES*C1);
  k_zero<<<blocks((long)NNODES*HEADS, TB), TB, 0, stream>>>(z1, NNODES*HEADS);
  k_edge<<<blocks((long)Etot*HEADS, TB), TB, 0, stream>>>(ei, E, NNODES, as1, ad1, ex1, z1, HEADS);
  k_agg<<<Etot, 128, 0, stream>>>(ei, E, NNODES, ex1, z1, xw1, out1, HEADS, C1);
  k_elu_pack<<<blocks((long)Mt*KS2*256, TB), TB, 0, stream>>>(out1, b1, apack2, Mt, KS2, C1);

  // ---- layer 2 ----
  k_gemm4<<<blocks((long)Mt*NtB2*32, TB), TB, 0, stream>>>((const v8u*)apack2, (const v8u*)bpack2,
                                                           xw2, Mt, NtB2, KS2, OUTD);
  k_alpha<<<blocks((long)NNODES*32, TB), TB, 0, stream>>>(xw2, a_src2, a_dst2,
                                                          as2, ad2, NNODES, 1, OUTD);
  k_zero<<<blocks((long)NNODES*OUTD, TB), TB, 0, stream>>>(out2, NNODES*OUTD);
  k_zero<<<blocks((long)NNODES, TB), TB, 0, stream>>>(z2, NNODES);
  k_edge<<<blocks((long)Etot, TB), TB, 0, stream>>>(ei, E, NNODES, as2, ad2, ex2, z2, 1);
  k_agg<<<Etot, 128, 0, stream>>>(ei, E, NNODES, ex2, z2, xw2, out2, 1, OUTD);

  // ---- global mean + bias + batch broadcast ----
  k_zero<<<1, 256, 0, stream>>>(gsum, 256);
  k_colsum<<<40, OUTD, 0, stream>>>(out2, gsum, NNODES, OUTD);
  const int B = out_size / OUTD;
  k_writeout<<<1, OUTD, 0, stream>>>(gsum, b2, out, B, OUTD, 1.0f / (float)NNODES);
}
